// SemanticCodebookLoss_16879221473755
// MI455X (gfx1250) — matrix operand order repaired and checked by reading, not compiled, not run on hardware
//
#include <hip/hip_runtime.h>

typedef __attribute__((ext_vector_type(16))) __bf16 v16bf;
typedef __attribute__((ext_vector_type(8)))  __bf16 v8bf;
typedef __attribute__((ext_vector_type(4)))  __bf16 v4bf;
typedef __attribute__((ext_vector_type(8)))  float  v8f;
typedef __attribute__((ext_vector_type(4)))  float  v4f;
typedef __attribute__((ext_vector_type(4)))  int    vint4;

#define VSZ 1024   // codebook size (K dim of GEMM)
#define DSZ 256    // embedding dim (N dim of GEMM)
#define MTILE 64   // rows per block

#if __has_builtin(__builtin_amdgcn_global_load_async_to_lds_b128)
  #define HAVE_ASYNC_LDS 1
#else
  #define HAVE_ASYNC_LDS 0
#endif

#if __has_builtin(__builtin_amdgcn_s_wait_asynccnt)
  #define WAIT_ASYNC0() __builtin_amdgcn_s_wait_asynccnt(0)
#else
  #define WAIT_ASYNC0() asm volatile("s_wait_asynccnt 0x0" ::: "memory")
#endif

__global__ void zero_out_kernel(float* out) { out[0] = 0.0f; }

// Pre-swizzle codebook (fp32 [V][D]) into per-lane WMMA B-fragment order (bf16).
// Layout: [kb(32)][ntile(16)][lane(32)][elem(16)]
// lane l, elem e  ->  K = kb*32 + (l>=16 ? 16 : 0) + e ,  n = ntile*16 + (l&15)
__global__ void swizzle_codebook_kernel(const float* __restrict__ cb,
                                        __bf16* __restrict__ bswz) {
    int dst = blockIdx.x * 256 + threadIdx.x;       // 0 .. V*D-1
    int e  = dst & 15;
    int l  = (dst >> 4) & 31;
    int nt = (dst >> 9) & 15;
    int kb = dst >> 13;
    int k = kb * 32 + ((l >> 4) << 4) + e;
    int n = nt * 16 + (l & 15);
    bswz[dst] = (__bf16)cb[k * DSZ + n];
}

#if HAVE_ASYNC_LDS
// Copy one 16 KB swizzled-B chunk global -> LDS with async DMA (no VGPR round trip).
// Builtin signature (from compiler diagnostic): param0 = int4 addrspace(1)*,
// param1 = int4 addrspace(3)*, then imm offset, imm cpol.
typedef __attribute__((address_space(1))) vint4 gvint4;
typedef __attribute__((address_space(3))) vint4 lvint4;

__device__ __forceinline__ void async_copy_chunk(const __bf16* gsrc, __bf16* lds_dst, int tid) {
    #pragma unroll
    for (int i = 0; i < 4; ++i) {
        const int eoff = (tid + i * 256) * 8;   // 8 bf16 = one 16B slot
        __builtin_amdgcn_global_load_async_to_lds_b128(
            (gvint4*)(gsrc + eoff), (lvint4*)(lds_dst + eoff), 0, 0);
    }
}
#endif

__launch_bounds__(256)
__global__ void fused_codebook_loss_kernel(const float* __restrict__ logits,
                                           const int* __restrict__ targets,
                                           const __bf16* __restrict__ bswz,
                                           float* __restrict__ out,
                                           float scale) {
    __shared__ __align__(16) __bf16 Albuf[MTILE * VSZ];     // 128 KB: (probs - onehot) bf16
#if HAVE_ASYNC_LDS
    __shared__ __align__(16) __bf16 Bbuf[2][16 * 32 * 16];  // 2 x 16 KB double buffer
#else
    __shared__ __align__(16) __bf16 Bbuf[1][16 * 32 * 16];  // 16 KB single buffer
#endif

    const int tid  = threadIdx.x;
    const int lane = tid & 31;
    const int w    = tid >> 5;
    const long long rowBase = (long long)blockIdx.x * MTILE;

    // ---------------- Phase 1: softmax (minus onehot) into LDS as bf16 ----------------
    // wave w handles rows w*8 .. w*8+7 ; each lane holds 32 floats of its row.
    for (int rr = 0; rr < 8; ++rr) {
        const int row = w * 8 + rr;
        const float* lp = logits + (rowBase + row) * (long long)VSZ;

        float x[32];
        #pragma unroll
        for (int j = 0; j < 8; ++j) {
            v4f t = *(const v4f*)(lp + j * 128 + lane * 4);
            #pragma unroll
            for (int c = 0; c < 4; ++c) x[j * 4 + c] = t[c];
        }

        float m = x[0];
        #pragma unroll
        for (int i = 1; i < 32; ++i) m = fmaxf(m, x[i]);
        #pragma unroll
        for (int o = 16; o > 0; o >>= 1) m = fmaxf(m, __shfl_xor(m, o, 32));

        float s = 0.0f;
        #pragma unroll
        for (int i = 0; i < 32; ++i) { x[i] = __expf(x[i] - m); s += x[i]; }
        #pragma unroll
        for (int o = 16; o > 0; o >>= 1) s += __shfl_xor(s, o, 32);

        const float inv = 1.0f / s;
        const int tgt = targets[rowBase + row];

        #pragma unroll
        for (int j = 0; j < 8; ++j) {
            const int col = j * 128 + lane * 4;
            v4bf pk;
            #pragma unroll
            for (int c = 0; c < 4; ++c) {
                float p = x[j * 4 + c] * inv;
                if (col + c == tgt) p -= 1.0f;   // fold in the -onehot(t)
                pk[c] = (__bf16)p;
            }
            *(v4bf*)&Albuf[row * VSZ + col] = pk;
        }
    }
    __syncthreads();

    // ---------------- Phase 2: WMMA GEMM  C[64x256] = A[64x1024] * B[1024x256] ----------------
    const int rowTile  = w & 3;        // which 16-row tile this wave owns
    const int colGroup = w >> 2;       // 0 or 1: col tiles 0-7 or 8-15

    v8f acc[8];
    #pragma unroll
    for (int j = 0; j < 8; ++j)
        #pragma unroll
        for (int i = 0; i < 8; ++i) acc[j][i] = 0.0f;

    const int arow = rowTile * 16 + (lane & 15);
    const int koff = (lane >> 4) << 4;           // lanes 16-31 take K+16..31

#if HAVE_ASYNC_LDS
    // -------- async double-buffered B staging: one barrier per K step --------
    async_copy_chunk(bswz, &Bbuf[0][0], tid);
    for (int kb = 0; kb < 32; ++kb) {
        const int c = kb & 1;
        WAIT_ASYNC0();        // my async writes into Bbuf[c] are done
        __syncthreads();      // all waves: Bbuf[c] filled AND Bbuf[1-c] no longer being read

        if (kb + 1 < 32) async_copy_chunk(bswz + (kb + 1) * 8192, &Bbuf[1 - c][0], tid);

        // A fragment from LDS in contiguous k order (elem s -> K base+s), same order as B
        const __bf16* ap = &Albuf[arow * VSZ + kb * 32 + koff];
        v8bf a0 = *(const v8bf*)(ap);        // K = base+0..7
        v8bf a1 = *(const v8bf*)(ap + 8);    // K = base+8..15
        v16bf a;
        #pragma unroll
        for (int i = 0; i < 8; ++i) { a[i] = a0[i]; a[i + 8] = a1[i]; }

        #pragma unroll
        for (int j = 0; j < 8; ++j) {
            const int nt = colGroup * 8 + j;
            v16bf b = *(const v16bf*)&Bbuf[c][(nt * 32 + lane) * 16];
            acc[j] = __builtin_amdgcn_wmma_f32_16x16x32_bf16(
                false, a, false, b, (short)0, acc[j], false, false);
        }
    }
#else
    // -------- fallback: synchronous single-buffer staging --------
    for (int kb = 0; kb < 32; ++kb) {
        {
            const float4* src = (const float4*)(bswz + kb * 8192);
            float4* dst = (float4*)&Bbuf[0][0];
            #pragma unroll
            for (int i = 0; i < 4; ++i) dst[tid + i * 256] = src[tid + i * 256];
        }
        __syncthreads();

        const __bf16* ap = &Albuf[arow * VSZ + kb * 32 + koff];
        v8bf a0 = *(const v8bf*)(ap);
        v8bf a1 = *(const v8bf*)(ap + 8);
        v16bf a;
        #pragma unroll
        for (int i = 0; i < 8; ++i) { a[i] = a0[i]; a[i + 8] = a1[i]; }

        #pragma unroll
        for (int j = 0; j < 8; ++j) {
            const int nt = colGroup * 8 + j;
            v16bf b = *(const v16bf*)&Bbuf[0][(nt * 32 + lane) * 16];
            acc[j] = __builtin_amdgcn_wmma_f32_16x16x32_bf16(
                false, a, false, b, (short)0, acc[j], false, false);
        }
        __syncthreads();
    }
#endif

    // ---------------- Phase 3: sum of squares + reduction ----------------
    float s2 = 0.0f;
    #pragma unroll
    for (int j = 0; j < 8; ++j)
        #pragma unroll
        for (int i = 0; i < 8; ++i) s2 += acc[j][i] * acc[j][i];

    #pragma unroll
    for (int o = 16; o > 0; o >>= 1) s2 += __shfl_xor(s2, o, 32);

    __syncthreads();                        // Albuf no longer needed; reuse for reduction
    float* warpSums = (float*)&Albuf[0];
    if (lane == 0) warpSums[w] = s2;
    __syncthreads();
    if (tid == 0) {
        float t = 0.0f;
        #pragma unroll
        for (int i = 0; i < 8; ++i) t += warpSums[i];
        atomicAdd(out, t * scale);
    }
}

extern "C" void kernel_launch(void* const* d_in, const int* in_sizes, int n_in,
                              void* d_out, int out_size, void* d_ws, size_t ws_size,
                              hipStream_t stream) {
    const float* logits  = (const float*)d_in[0];   // [N, V] fp32 (NHWC-flat)
    const int*   targets = (const int*)d_in[1];     // [N]
    const float* cb      = (const float*)d_in[2];   // [V, D] fp32
    float* out = (float*)d_out;
    __bf16* bswz = (__bf16*)d_ws;                   // 512 KB swizzled bf16 codebook

    const int N = in_sizes[1];                      // 65536 samples
    const float scale = 1.0f / ((float)N * (float)DSZ);

    zero_out_kernel<<<1, 1, 0, stream>>>(out);
    swizzle_codebook_kernel<<<(VSZ * DSZ) / 256, 256, 0, stream>>>(cb, bswz);
    fused_codebook_loss_kernel<<<N / MTILE, 256, 0, stream>>>(logits, targets, bswz, out, scale);
}
